// CNF_55379308314905
// MI455X (gfx1250) — matrix operand order repaired and checked by reading, not compiled, not run on hardware
//
#include <hip/hip_runtime.h>
#include <hip/hip_bf16.h>

typedef __attribute__((ext_vector_type(16))) _Float16 v16h;
typedef __attribute__((ext_vector_type(8)))  _Float16 v8h;
typedef __attribute__((ext_vector_type(8)))  float    v8f;

#define DIMD   32      // ODE state dim
#define NCOND  8
#define NYD    9       // t + cond
#define NWID   128     // hidden width
#define SBLK   4       // samples per block
#define NTIL   9       // N tiles: (SBLK*32 + SBLK h-cols -> pad) / 16
#define JCOLS  (NTIL*16)   // 144 logical columns (128 J + 4 h + 12 pad)
#define JSTR   144         // halfs per column (128 rows + 16 pad) -> 288B, 32B-aligned
#define NSTP   20
#define DTH    (-0.05f)

__global__ __launch_bounds__(256) void cnf_heun_wmma(
    const float* __restrict__ x,     const float* __restrict__ cond,
    const float* __restrict__ l0_W1, const float* __restrict__ l0_b1,
    const float* __restrict__ l0_W2, const float* __restrict__ l0_b2,
    const float* __restrict__ l0_W3,
    const float* __restrict__ mid_W1, const float* __restrict__ mid_b1,
    const float* __restrict__ mid_W2, const float* __restrict__ mid_b2,
    const float* __restrict__ mid_W3,
    const float* __restrict__ lN_W1, const float* __restrict__ lN_b1,
    const float* __restrict__ lN_W2, const float* __restrict__ lN_b2,
    const float* __restrict__ lN_W3,
    float* __restrict__ out)
{
    // ---- LDS (~60 KB) ----
    __shared__ __attribute__((aligned(32))) _Float16 Jbuf[JCOLS * JSTR]; // 41472 B
    __shared__ float u_lds[NWID * SBLK];        // per-row, per-sample Jacobian scale
    __shared__ float sg_mid[2][NWID * SBLK];    // sigmoid gates, mid layers
    __shared__ float w3_mid[2][NWID * SBLK];    // W3@ty, mid layers
    __shared__ float s0g[NWID * SBLK], w30g[NWID * SBLK];
    __shared__ float sNg[DIMD * SBLK], w3Ng[DIMD * SBLK];
    __shared__ float psi[SBLK * DIMD], psi2[SBLK * DIMD];
    __shared__ float kf[2][SBLK * DIMD];
    __shared__ float trp[SBLK * DIMD];
    __shared__ float ktr[2][SBLK];
    __shared__ float logp[SBLK];
    __shared__ float tyv[SBLK * NYD];

    const int tid  = threadIdx.x;
    const int lane = tid & 31;
    const int w    = tid >> 5;        // wave 0..7 == M-tile of mid GEMM
    const int hi   = lane >> 4;       // half-wave select
    const int ln   = lane & 15;       // N / M position within tile
    const int g0   = blockIdx.x * SBLK;

    // ---- Preload WMMA A-fragments (row-major f32 -> contiguous-K 16-bit A layout, f16) ----
    // A 16x32 layout: lanes: M = lane%16; half e -> K = 32*kc + 16*hi + e
    v16h Amid[2][4];
    {
        const int m = w * 16 + ln;
        for (int L = 0; L < 2; ++L) {
            const float* Wp = mid_W1 + (L * NWID + m) * NWID;
            for (int c = 0; c < 4; ++c) {
                const int kb = 32 * c + 16 * hi;
                v16h a;
                #pragma unroll
                for (int e = 0; e < 8; ++e) a[e]     = (_Float16)Wp[kb + e];
                #pragma unroll
                for (int e = 0; e < 8; ++e) a[e + 8] = (_Float16)Wp[kb + 8 + e];
                Amid[L][c] = a;
            }
        }
    }
    v16h AlN[4];   // last layer: wave parity picks M-tile (rows 0-15 / 16-31)
    {
        const int m = (w & 1) * 16 + ln;
        const float* Wp = lN_W1 + m * NWID;
        for (int c = 0; c < 4; ++c) {
            const int kb = 32 * c + 16 * hi;
            v16h a;
            #pragma unroll
            for (int e = 0; e < 8; ++e) a[e]     = (_Float16)Wp[kb + e];
            #pragma unroll
            for (int e = 0; e < 8; ++e) a[e + 8] = (_Float16)Wp[kb + 8 + e];
            AlN[c] = a;
        }
    }

    if (tid < SBLK * DIMD) psi[tid] = x[g0 * DIMD + tid];
    if (tid < SBLK) logp[tid] = 0.0f;
    __syncthreads();

    for (int step = 0; step < NSTP; ++step) {
        const float t1 = 1.0f + DTH * (float)step;
        for (int stage = 0; stage < 2; ++stage) {
            const float t = stage ? (t1 + DTH) : t1;
            const float* psrc = stage ? psi2 : psi;

            // ---- gates: sigmoid(W2@ty+b2) and W3@ty for every layer ----
            if (tid < SBLK * NYD) {
                const int s = tid / NYD, j = tid - s * NYD;
                tyv[tid] = (j == 0) ? t : cond[(g0 + s) * NCOND + (j - 1)];
            }
            __syncthreads();
            for (int j = tid; j < (3 * NWID + DIMD) * SBLK; j += 256) {
                const int q = j >> 2, s = j & (SBLK - 1);
                const float *W2p, *W3p; float b2v; float *sgp, *w3p;
                if (q < NWID) {
                    W2p = l0_W2 + q * NYD; W3p = l0_W3 + q * NYD; b2v = l0_b2[q];
                    sgp = &s0g[q * SBLK + s]; w3p = &w30g[q * SBLK + s];
                } else if (q < 3 * NWID) {
                    const int L = (q - NWID) >> 7, m = (q - NWID) & 127;
                    W2p = mid_W2 + (L * NWID + m) * NYD; W3p = mid_W3 + (L * NWID + m) * NYD;
                    b2v = mid_b2[L * NWID + m];
                    sgp = &sg_mid[L][m * SBLK + s]; w3p = &w3_mid[L][m * SBLK + s];
                } else {
                    const int m = q - 3 * NWID;
                    W2p = lN_W2 + m * NYD; W3p = lN_W3 + m * NYD; b2v = lN_b2[m];
                    sgp = &sNg[m * SBLK + s]; w3p = &w3Ng[m * SBLK + s];
                }
                float a2 = b2v, a3 = 0.0f;
                #pragma unroll
                for (int k = 0; k < NYD; ++k) {
                    const float tv = tyv[s * NYD + k];
                    a2 = fmaf(W2p[k], tv, a2); a3 = fmaf(W3p[k], tv, a3);
                }
                *sgp = 1.0f / (1.0f + __expf(-a2));
                *w3p = a3;
            }
            __syncthreads();

            // ---- layer 0: h0 = tanh((W10 psi + b)*g + w3ty); J0 = diag(u0) W10 ----
            for (int j = tid; j < NWID * SBLK; j += 256) {
                const int m = j >> 2, s = j & 3;
                const float* Wp = l0_W1 + m * DIMD;
                float acc = l0_b1[m];
                #pragma unroll
                for (int k = 0; k < DIMD; ++k) acc = fmaf(Wp[k], psrc[s * DIMD + k], acc);
                const float g = s0g[m * SBLK + s];
                const float a = tanhf(acc * g + w30g[m * SBLK + s]);
                Jbuf[(128 + s) * JSTR + m] = (_Float16)a;       // h column
                u_lds[m * SBLK + s] = g * (1.0f - a * a);
            }
            __syncthreads();
            for (int j = tid; j < NWID * NWID; j += 256) {
                const int col = j & 127, m = j >> 7;
                const int s = col >> 5, k = col & 31;
                Jbuf[col * JSTR + m] = (_Float16)(u_lds[m * SBLK + s] * l0_W1[m * DIMD + k]);
            }
            __syncthreads();

            // ---- mid layers: [Jnext|Whcol] = W @ [J|h], then gate/scale ----
            for (int L = 0; L < 2; ++L) {
                v8f acc[NTIL];
                #pragma unroll
                for (int n = 0; n < NTIL; ++n) {
                    v8f c = {};
                    #pragma unroll
                    for (int kc = 0; kc < 4; ++kc) {
                        // B 32x16 layout: lane = N; halfs = 16 contiguous K (32B aligned)
                        const v16h b = *(const v16h*)&Jbuf[(n * 16 + ln) * JSTR + 32 * kc + 16 * hi];
                        c = __builtin_amdgcn_wmma_f32_16x16x32_f16(
                                false, Amid[L][kc], false, b, (short)0, c, false, false);
                    }
                    acc[n] = c;
                }
                __syncthreads();                 // all J reads complete before rewrites
                if (ln < SBLK) {                 // h columns: activation + new scale u
                    const int s = ln;
                    #pragma unroll
                    for (int r = 0; r < 8; ++r) {
                        const int m = w * 16 + 8 * hi + r;
                        const float val = acc[8][r] + mid_b1[L * NWID + m];
                        const float g = sg_mid[L][m * SBLK + s];
                        const float a = tanhf(val * g + w3_mid[L][m * SBLK + s]);
                        Jbuf[(128 + s) * JSTR + m] = (_Float16)a;
                        u_lds[m * SBLK + s] = g * (1.0f - a * a);
                    }
                }
                __syncthreads();                 // u ready
                #pragma unroll
                for (int n = 0; n < 8; ++n) {    // scale + store J tiles (f32 -> f16)
                    const int col = n * 16 + ln;
                    const int s = col >> 5;
                    const int mb = w * 16 + 8 * hi;
                    v8h o;
                    #pragma unroll
                    for (int r = 0; r < 8; ++r)
                        o[r] = (_Float16)(acc[n][r] * u_lds[(mb + r) * SBLK + s]);
                    *(v8h*)&Jbuf[col * JSTR + mb] = o;
                }
                __syncthreads();
            }

            // ---- last layer GEMM: Y(32 x 144) = W1N @ [J|h] ----
            v8f yacc[3]; int ynt[3]; int ycnt = 0;
            for (int j = w; j < 2 * NTIL; j += 8) {   // j&1 == w&1 -> matches AlN M-tile
                const int nt = j >> 1;
                v8f c = {};
                #pragma unroll
                for (int kc = 0; kc < 4; ++kc) {
                    const v16h b = *(const v16h*)&Jbuf[(nt * 16 + ln) * JSTR + 32 * kc + 16 * hi];
                    c = __builtin_amdgcn_wmma_f32_16x16x32_f16(
                            false, AlN[kc], false, b, (short)0, c, false, false);
                }
                yacc[ycnt] = c; ynt[ycnt] = nt; ++ycnt;
            }
            __syncthreads();                     // J reads done; reuse Jbuf as f32 Y
            float* Ybuf = (float*)Jbuf;          // Y[col*33 + i], i = output row 0..31
            for (int tt = 0; tt < ycnt; ++tt) {
                const int col = ynt[tt] * 16 + ln;
                const int ib = (w & 1) * 16 + 8 * hi;
                #pragma unroll
                for (int r = 0; r < 8; ++r) Ybuf[col * 33 + ib + r] = yacc[tt][r];
            }
            __syncthreads();

            // ---- finish: vector field f and Jacobian trace ----
            if (tid < SBLK * DIMD) {
                const int s = tid >> 5, i = tid & 31;
                const float y = Ybuf[(128 + s) * 33 + i];
                kf[stage][tid] = (y + lN_b1[i]) * sNg[i * SBLK + s] + w3Ng[i * SBLK + s];
                trp[tid] = sNg[i * SBLK + s] * Ybuf[(s * 32 + i) * 33 + i]; // diagonal pick
            }
            __syncthreads();
            if (tid < SBLK) {
                float tr = 0.0f;
                for (int i = 0; i < DIMD; ++i) tr += trp[tid * DIMD + i];
                ktr[stage][tid] = tr;
            }
            if (stage == 0) {
                if (tid < SBLK * DIMD) psi2[tid] = psi[tid] + DTH * kf[0][tid];
            }
            __syncthreads();
        } // stage

        // Heun combine
        if (tid < SBLK * DIMD) psi[tid] += 0.5f * DTH * (kf[0][tid] + kf[1][tid]);
        if (tid < SBLK)        logp[tid] += 0.5f * DTH * (ktr[0][tid] + ktr[1][tid]);
        __syncthreads();
    } // step

    if (tid < SBLK) {
        float ss = 0.0f;
        for (int i = 0; i < DIMD; ++i) { const float v = psi[tid * DIMD + i]; ss += v * v; }
        const float LOG2PI = 1.8378770664093453f;
        out[g0 + tid] = -0.5f * ((float)DIMD * LOG2PI + ss) + logp[tid];
    }
}

extern "C" void kernel_launch(void* const* d_in, const int* in_sizes, int n_in,
                              void* d_out, int out_size, void* d_ws, size_t ws_size,
                              hipStream_t stream) {
    const float* x      = (const float*)d_in[0];
    const float* cond   = (const float*)d_in[1];
    const float* l0_W1  = (const float*)d_in[2];
    const float* l0_b1  = (const float*)d_in[3];
    const float* l0_W2  = (const float*)d_in[4];
    const float* l0_b2  = (const float*)d_in[5];
    const float* l0_W3  = (const float*)d_in[6];
    const float* mid_W1 = (const float*)d_in[7];
    const float* mid_b1 = (const float*)d_in[8];
    const float* mid_W2 = (const float*)d_in[9];
    const float* mid_b2 = (const float*)d_in[10];
    const float* mid_W3 = (const float*)d_in[11];
    const float* lN_W1  = (const float*)d_in[12];
    const float* lN_b1  = (const float*)d_in[13];
    const float* lN_W2  = (const float*)d_in[14];
    const float* lN_b2  = (const float*)d_in[15];
    const float* lN_W3  = (const float*)d_in[16];

    const int nb   = in_sizes[0] / DIMD;   // batch (4096)
    const int grid = nb / SBLK;            // 1024 workgroups

    cnf_heun_wmma<<<grid, 256, 0, stream>>>(
        x, cond, l0_W1, l0_b1, l0_W2, l0_b2, l0_W3,
        mid_W1, mid_b1, mid_W2, mid_b2, mid_W3,
        lN_W1, lN_b1, lN_W2, lN_b2, lN_W3,
        (float*)d_out);
}
